// SemanticDistanceFusion_6073083757181
// MI455X (gfx1250) — matrix operand order repaired and checked by reading, not compiled, not run on hardware
//
#include <hip/hip_runtime.h>
#include <math.h>

// ---------------------------------------------------------------------------
// Problem constants (from reference): B=8, N=256, D=512, H=2048, M = B*N = 2048
// ---------------------------------------------------------------------------

typedef __attribute__((ext_vector_type(16))) __bf16 v16bf;
typedef __attribute__((ext_vector_type(8)))  float  v8f;
typedef unsigned int u32x4 __attribute__((ext_vector_type(4)));

union FragBF { u32x4 q[2]; v16bf v; };   // 32 bytes: 16 bf16 per lane

__device__ __forceinline__ unsigned short f2bf(float f) {
  unsigned int u = __float_as_uint(f);
  u += 0x7FFFu + ((u >> 16) & 1u);       // round-to-nearest-even
  return (unsigned short)(u >> 16);
}

__device__ __forceinline__ v8f wmma_bf16(v16bf a, v16bf b, v8f c) {
  // (neg_a, A, neg_b, B, c_mod, C, reuse_a, reuse_b)
  return __builtin_amdgcn_wmma_f32_16x16x32_bf16(false, a, false, b, (short)0, c,
                                                 false, false);
}

// ---------------------------------------------------------------------------
// Kernel 1: pairwise L1 distance  md[b,i,j] = sum_d |xv[b,i,d] - xa[b,j,d]|
// grid = B * (N/8) = 256 blocks, 256 threads; thread t owns column j = t,
// block owns 8 consecutive i rows (amortizes xa traffic 8x).
// ---------------------------------------------------------------------------
__global__ void dist_kernel(const float* __restrict__ xv,
                            const float* __restrict__ xa,
                            float* __restrict__ md) {
  const int tid = threadIdx.x;
  const int b   = blockIdx.x >> 5;        // 32 blocks per batch
  const int i0  = (blockIdx.x & 31) * 8;

  __shared__ float xvs[8 * 512];          // 16 KB
  __shared__ float xas[256 * 17];         // 17 KB, pitch 17 -> conflict-free

  for (int e = tid; e < 8 * 512; e += 256)
    xvs[e] = xv[(size_t)(b * 256 + i0) * 512 + e];

  float acc[8];
#pragma unroll
  for (int ii = 0; ii < 8; ++ii) acc[ii] = 0.f;

  for (int d0 = 0; d0 < 512; d0 += 16) {
    __syncthreads();
#pragma unroll
    for (int it = 0; it < 16; ++it) {     // coalesced: consecutive e -> consecutive d
      int e = tid + it * 256;
      int row = e >> 4, dd = e & 15;
      xas[row * 17 + dd] = xa[(size_t)(b * 256 + row) * 512 + d0 + dd];
    }
    __syncthreads();
#pragma unroll
    for (int dd = 0; dd < 16; ++dd) {
      float av = xas[tid * 17 + dd];      // lane-per-row, distinct banks
#pragma unroll
      for (int ii = 0; ii < 8; ++ii)
        acc[ii] += fabsf(xvs[ii * 512 + d0 + dd] - av);
    }
  }
#pragma unroll
  for (int ii = 0; ii < 8; ++ii)
    md[(size_t)(b * 256 + i0 + ii) * 256 + tid] = acc[ii];
}

// ---------------------------------------------------------------------------
// Kernel 2: row means (md_v) and column means (md_a), deterministic reductions.
// grid = 2 * 2048 blocks of 256 threads.
// ---------------------------------------------------------------------------
__global__ void reduce_md(const float* __restrict__ md,
                          float* __restrict__ md_v,
                          float* __restrict__ md_a) {
  const int tid = threadIdx.x;
  const int idx = blockIdx.x;
  float v;
  if (idx < 2048) {
    v = md[(size_t)idx * 256 + tid];                       // reduce over j
  } else {
    int r = idx - 2048;
    int b = r >> 8, j = r & 255;
    v = md[(size_t)(b * 256 + tid) * 256 + j];             // reduce over i
  }
  __shared__ float red[256];
  red[tid] = v;
  __syncthreads();
  for (int s = 128; s > 0; s >>= 1) {
    if (tid < s) red[tid] += red[tid + s];
    __syncthreads();
  }
  if (tid == 0) {
    float m = red[0] * (1.0f / 256.0f);
    if (idx < 2048) md_v[idx] = m;
    else            md_a[idx - 2048] = m;
  }
}

// ---------------------------------------------------------------------------
// Kernel 3: x * rowscale -> bf16  (xvw / xaw activation staging)
// ---------------------------------------------------------------------------
__global__ void scale_to_bf16(const float* __restrict__ x,
                              const float* __restrict__ s,
                              unsigned short* __restrict__ o, int n) {
  int i = blockIdx.x * 256 + threadIdx.x;
  if (i < n) o[i] = f2bf(x[i] * s[i >> 9]);   // D = 512 = 2^9
}

// ---------------------------------------------------------------------------
// Kernel 4: fp32 -> bf16 transpose (weights stored N-major so WMMA B-fragments
// are contiguous per lane). Classic 32x32 LDS tile transpose.
// ---------------------------------------------------------------------------
__global__ void transpose_bf16(const float* __restrict__ src,
                               unsigned short* __restrict__ dst,
                               int rows, int cols) {
  __shared__ float tile[32][33];
  int x = blockIdx.x * 32 + threadIdx.x;
  int y = blockIdx.y * 32 + threadIdx.y;
#pragma unroll
  for (int i = 0; i < 32; i += 8)
    tile[threadIdx.y + i][threadIdx.x] = src[(size_t)(y + i) * cols + x];
  __syncthreads();
  int xo = blockIdx.y * 32 + threadIdx.x;
  int yo = blockIdx.x * 32 + threadIdx.y;
#pragma unroll
  for (int i = 0; i < 32; i += 8)
    dst[(size_t)(yo + i) * rows + xo] = f2bf(tile[threadIdx.x][threadIdx.y + i]);
}

// ---------------------------------------------------------------------------
// Kernel 5: bf16 WMMA GEMM.  C[m][n] = act( sum_k A[m][k]*Bt[n][k] + bias[n] )
// A: M x K bf16 row-major, Bt: N x K bf16 (transposed weights).
// Workgroup: 256 threads = 8 waves, tile 128(M) x 128(N), BK = 32.
// Wave (wm in 0..3, wn in 0..1) owns a 32x64 patch = 2x4 WMMA tiles
// -> 8 v_wmma per K-step against 6 ds_load_b128-pair fragment loads.
// Double-buffered LDS, ONE barrier per K-step: stage tile k+1 into buf^1
// after the barrier while computing from buf (reads of buf^1 from iteration
// i-1 are complete before any wave passes barrier i).
// LDS row pitch = 5 * uint4 (64B data + 16B pad) -> aligned b128 frag loads.
// MODE: 0 = exact-GELU -> bf16, 1 = identity -> bf16, 2 = identity -> f32
// ---------------------------------------------------------------------------
template <int MODE>
__global__ void gemm_kernel(const unsigned short* __restrict__ A,
                            const unsigned short* __restrict__ Bt,
                            const float* __restrict__ bias,
                            void* __restrict__ Cptr,
                            int K, int ldC) {
  __shared__ u32x4 sa[2 * 128 * 5];   // 2 x 10 KB : A tiles 128 x 32 bf16 (padded)
  __shared__ u32x4 sb[2 * 128 * 5];   // 2 x 10 KB : Bt tiles 128 x 32 bf16 (padded)

  const int tid  = threadIdx.x;
  const int lane = tid & 31;
  const int wave = tid >> 5;
  const int half = lane >> 4;     // 0 / 1  (lane groups per ISA layout)
  const int lrow = lane & 15;
  const int wm   = wave & 3;      // 4 waves along M (4 x 32 = 128)
  const int wn   = wave >> 2;     // 2 waves along N (2 x 64 = 128)
  const int m0   = blockIdx.y * 128;
  const int n0   = blockIdx.x * 128;

  // Stage one 128x32 A tile and 128x32 Bt tile (1024 x 16B chunks, 4/thread)
  auto stage = [&](int buf, int kt) {
    int base = buf * 640;
#pragma unroll
    for (int h = 0; h < 2; ++h) {
      int c = tid + h * 256;
      int r = c >> 2, q = c & 3;
      sa[base + r * 5 + q] = *(const u32x4*)(A + (size_t)(m0 + r) * K + kt + q * 8);
      sb[base + r * 5 + q] = *(const u32x4*)(Bt + (size_t)(n0 + r) * K + kt + q * 8);
    }
  };

  v8f zero = {};
  v8f acc[2][4];
#pragma unroll
  for (int ti = 0; ti < 2; ++ti)
#pragma unroll
    for (int tj = 0; tj < 4; ++tj) acc[ti][tj] = zero;

  stage(0, 0);
  int buf = 0;
  for (int kt = 0; kt < K; kt += 32, buf ^= 1) {
    __syncthreads();
    if (kt + 32 < K) stage(buf ^ 1, kt + 32);

    const int base = buf * 640;
    // A fragments (16x32): lanes 0-15 K{0..7,16..23}, lanes 16-31 K{8..15,24..31}
    FragBF af[2];
#pragma unroll
    for (int ti = 0; ti < 2; ++ti) {
      int r = wm * 32 + ti * 16 + lrow;
      af[ti].q[0] = sa[base + r * 5 + half];
      af[ti].q[1] = sa[base + r * 5 + 2 + half];
    }
    // B fragments (32x16): lanes 0-15 K{0..7,16..23}, lanes 16-31 K{8..15,24..31}
    FragBF bfr[4];
#pragma unroll
    for (int tj = 0; tj < 4; ++tj) {
      int r = wn * 64 + tj * 16 + lrow;
      bfr[tj].q[0] = sb[base + r * 5 + half];
      bfr[tj].q[1] = sb[base + r * 5 + 2 + half];
    }
#pragma unroll
    for (int ti = 0; ti < 2; ++ti)
#pragma unroll
      for (int tj = 0; tj < 4; ++tj)
        acc[ti][tj] = wmma_bf16(af[ti].v, bfr[tj].v, acc[ti][tj]);
  }

  // Epilogue: C/D layout -> lane element v is C[half*8 + v][lrow] of a 16x16 tile
#pragma unroll
  for (int ti = 0; ti < 2; ++ti) {
#pragma unroll
    for (int tj = 0; tj < 4; ++tj) {
      int col = n0 + wn * 64 + tj * 16 + lrow;
      float bv = bias[col];
#pragma unroll
      for (int v = 0; v < 8; ++v) {
        int row = m0 + wm * 32 + ti * 16 + half * 8 + v;
        float val = acc[ti][tj][v] + bv;
        if (MODE == 0)
          val = 0.5f * val * (1.0f + erff(val * 0.70710678118654752f));
        if (MODE == 2)
          ((float*)Cptr)[(size_t)row * ldC + col] = val;
        else
          ((unsigned short*)Cptr)[(size_t)row * ldC + col] = f2bf(val);
      }
    }
  }
}

// ---------------------------------------------------------------------------
// Kernel 6: gating.  gates = [xmm,dist] @ Wfuse + bfuse (1024x2 GEMV per row),
// out = sigmoid(g0)*xmm + tanh(g1)*dist.  One block per row.
// ---------------------------------------------------------------------------
__global__ void fuse_kernel(const float* __restrict__ xmm,
                            const float* __restrict__ dist,
                            const float* __restrict__ Wf,
                            const float* __restrict__ bf,
                            float* __restrict__ out) {
  const int tid = threadIdx.x;
  const int r   = blockIdx.x;
  const float* xr = xmm  + (size_t)r * 512;
  const float* dr = dist + (size_t)r * 512;

  float g0 = 0.f, g1 = 0.f;
#pragma unroll
  for (int it = 0; it < 2; ++it) {
    int c = tid + it * 256;
    float xc = xr[c], dc = dr[c];
    g0 += xc * Wf[c * 2 + 0] + dc * Wf[(512 + c) * 2 + 0];
    g1 += xc * Wf[c * 2 + 1] + dc * Wf[(512 + c) * 2 + 1];
  }
  __shared__ float r0[256], r1[256];
  r0[tid] = g0; r1[tid] = g1;
  __syncthreads();
  for (int s = 128; s > 0; s >>= 1) {
    if (tid < s) { r0[tid] += r0[tid + s]; r1[tid] += r1[tid + s]; }
    __syncthreads();
  }
  float gs = 1.0f / (1.0f + expf(-(r0[0] + bf[0])));
  float gt = tanhf(r1[0] + bf[1]);
#pragma unroll
  for (int it = 0; it < 2; ++it) {
    int c = tid + it * 256;
    out[(size_t)r * 512 + c] = gs * xr[c] + gt * dr[c];
  }
}

// ---------------------------------------------------------------------------
// Launcher
// ---------------------------------------------------------------------------
extern "C" void kernel_launch(void* const* d_in, const int* in_sizes, int n_in,
                              void* d_out, int out_size, void* d_ws, size_t ws_size,
                              hipStream_t stream) {
  (void)in_sizes; (void)n_in; (void)out_size; (void)ws_size;
  const float* xmm   = (const float*)d_in[0];
  const float* xv    = (const float*)d_in[1];
  const float* xa    = (const float*)d_in[2];
  const float* W1v   = (const float*)d_in[3];
  const float* b1v   = (const float*)d_in[4];
  const float* W1a   = (const float*)d_in[5];
  const float* b1a   = (const float*)d_in[6];
  const float* W2v   = (const float*)d_in[7];
  const float* b2v   = (const float*)d_in[8];
  const float* W2a   = (const float*)d_in[9];
  const float* b2a   = (const float*)d_in[10];
  const float* Wout  = (const float*)d_in[11];
  const float* bout  = (const float*)d_in[12];
  const float* Wfuse = (const float*)d_in[13];
  const float* bfuse = (const float*)d_in[14];

  char* ws = (char*)d_ws;
  size_t off = 0;
  auto alloc = [&](size_t bytes) -> void* {
    void* p = ws + off;
    off = (off + bytes + 255) & ~(size_t)255;
    return p;
  };
  float*          md     = (float*)alloc(2048ull * 256 * 4);          // 2 MB
  float*          md_v   = (float*)alloc(2048ull * 4);
  float*          md_a   = (float*)alloc(2048ull * 4);
  unsigned short* xvw    = (unsigned short*)alloc(2048ull * 512 * 2); // 2 MB
  unsigned short* xaw    = (unsigned short*)alloc(2048ull * 512 * 2);
  unsigned short* W1vT   = (unsigned short*)alloc(2048ull * 512 * 2);
  unsigned short* W1aT   = (unsigned short*)alloc(2048ull * 512 * 2);
  unsigned short* W2vT   = (unsigned short*)alloc(512ull * 2048 * 2);
  unsigned short* W2aT   = (unsigned short*)alloc(512ull * 2048 * 2);
  unsigned short* WoutT  = (unsigned short*)alloc(512ull * 1024 * 2); // 1 MB
  unsigned short* Hbuf   = (unsigned short*)alloc(2048ull * 2048 * 2);// 8 MB (reused v then a)
  unsigned short* concat = (unsigned short*)alloc(2048ull * 1024 * 2);// 4 MB
  float*          distb  = (float*)alloc(2048ull * 512 * 4);          // 4 MB

  // Phase 1: L1 distance + means + bf16 activation staging
  dist_kernel<<<256, 256, 0, stream>>>(xv, xa, md);
  reduce_md<<<4096, 256, 0, stream>>>(md, md_v, md_a);
  scale_to_bf16<<<4096, 256, 0, stream>>>(xv, md_v, xvw, 2048 * 512);
  scale_to_bf16<<<4096, 256, 0, stream>>>(xa, md_a, xaw, 2048 * 512);

  // Weight staging: fp32 -> bf16, transposed (N-major for WMMA B fragments)
  transpose_bf16<<<dim3(64, 16), dim3(32, 8), 0, stream>>>(W1v, W1vT, 512, 2048);
  transpose_bf16<<<dim3(64, 16), dim3(32, 8), 0, stream>>>(W1a, W1aT, 512, 2048);
  transpose_bf16<<<dim3(16, 64), dim3(32, 8), 0, stream>>>(W2v, W2vT, 2048, 512);
  transpose_bf16<<<dim3(16, 64), dim3(32, 8), 0, stream>>>(W2a, W2aT, 2048, 512);
  transpose_bf16<<<dim3(16, 32), dim3(32, 8), 0, stream>>>(Wout, WoutT, 1024, 512);

  // Phase 2: WMMA GEMM chain (stream order serializes Hbuf reuse)
  // grids: x = N/128, y = M/128
  gemm_kernel<0><<<dim3(16, 16), 256, 0, stream>>>(xvw, W1vT, b1v, Hbuf, 512, 2048);
  gemm_kernel<1><<<dim3(4, 16), 256, 0, stream>>>(Hbuf, W2vT, b2v, concat, 2048, 1024);
  gemm_kernel<0><<<dim3(16, 16), 256, 0, stream>>>(xaw, W1aT, b1a, Hbuf, 512, 2048);
  gemm_kernel<1><<<dim3(4, 16), 256, 0, stream>>>(Hbuf, W2aT, b2a, concat + 512, 2048, 1024);
  gemm_kernel<2><<<dim3(4, 16), 256, 0, stream>>>(concat, WoutT, bout, distb, 1024, 512);

  // Phase 3: gated fusion
  fuse_kernel<<<2048, 256, 0, stream>>>(xmm, distb, Wfuse, bfuse, (float*)d_out);
}
